// PointSetPooling_72078141161555
// MI455X (gfx1250) — matrix operand order repaired and checked by reading, not compiled, not run on hardware
//
#include <hip/hip_runtime.h>
#include <hip/hip_bf16.h>
#include <stdint.h>

typedef __attribute__((ext_vector_type(16))) _Float16 v16h;
typedef __attribute__((ext_vector_type(8)))  float    v8f;
typedef __attribute__((ext_vector_type(4)))  unsigned int u32x4;

#define N_POINTS_C   100000
#define N_KEYPOINTS_C 16384
#define N_PAIRS_C    524288
#define AGG_W        320      // padded K for output MLP (300 -> 320)
#define NPAD_OUT     304      // padded N for 300-wide outputs

// ---- workspace layout (dword offsets for swizzled f16 weight fragments) ----
#define W1F_DW 0               // 1 Kc * 2 Nt  * 256 = 512 dwords   (4x32)
#define W2F_DW 512             // 1 * 4  * 256 = 1024               (32x64)
#define W3F_DW 1536            // 2 * 8  * 256 = 4096               (64x128)
#define W4F_DW 5632            // 4 * 19 * 256 = 19456              (128x300)
#define O1F_DW 25088           // 10 * 19 * 256 = 48640             (300x300, K padded 320)
#define O2F_DW 73728           // 10 * 19 * 256 = 48640
#define WFRAG_TOTAL_DW 122368
#define AGG32_BYTE ((size_t)WFRAG_TOTAL_DW * 4)                  // 489472 (16B aligned)
#define AGG32_SZ   ((size_t)N_KEYPOINTS_C * AGG_W * 4)           // 20 MB f32
#define AGG16_BYTE (AGG32_BYTE + AGG32_SZ)
#define AGG16_SZ   ((size_t)N_KEYPOINTS_C * AGG_W * 2)           // 10 MB f16
#define Y1_BYTE    (AGG16_BYTE + AGG16_SZ)
#define Y1_SZ      ((size_t)N_KEYPOINTS_C * AGG_W * 2)           // 10 MB f16
#define ZERO_BYTES (AGG32_SZ + AGG16_SZ + Y1_SZ)

union HFrag { v16h h; unsigned int u[8]; u32x4 q[2]; };
union CFrag { v8f  v; float f[8]; };

__device__ __forceinline__ unsigned int pack2h(float a, float b) {
  union { _Float16 h[2]; unsigned int u; } r;
  r.h[0] = (_Float16)a; r.h[1] = (_Float16)b;
  return r.u;
}

// A-matrix (16x32 f16) K index for VGPR v, lane-half h  (ISA 7.12.2)
__device__ __forceinline__ int a_kin(int v, int h) {
  return (v < 4) ? (h * 8 + 2 * v) : (16 + h * 8 + 2 * (v - 4));
}
// B-matrix (32x16 f16) K index for VGPR v, lane-half h  (ISA 7.12.2 / 7.12.4)
__device__ __forceinline__ int b_kin(int v, int h) {
  return (v < 4) ? (h * 8 + 2 * v) : (16 + h * 8 + 2 * (v - 4));
}

// async global->LDS fill, CDNA5 GLOBAL_LOAD_ASYNC_TO_LDS_B128 (GVS mode), ASYNCcnt
__device__ __forceinline__ void async_copy_b128(unsigned lds_byte_off, unsigned gbl_byte_off,
                                                const void* sbase) {
  asm volatile("global_load_async_to_lds_b128 %0, %1, %2"
               :: "v"(lds_byte_off), "v"(gbl_byte_off), "s"(sbase)
               : "memory");
}
__device__ __forceinline__ void wait_async0() {
  asm volatile("s_wait_asynccnt 0x0" ::: "memory");
}
// LDS byte offset of a __shared__ object (generic LDS addr low 32 bits = offset)
__device__ __forceinline__ unsigned lds_off(const void* p) {
  return (unsigned)(uintptr_t)p;
}

__device__ __forceinline__ HFrag load_a_frag_lds(const _Float16* stage, int W, int kBase, int lane) {
  HFrag a;
  int m = lane & 15, h = lane >> 4;
#pragma unroll
  for (int v = 0; v < 8; ++v) {
    int k = kBase + a_kin(v, h);                       // even -> 4B aligned
    a.u[v] = *(const unsigned int*)(stage + m * W + k);
  }
  return a;
}

// fragments stored lane-major: dword index = tile*256 + lane*8 + v  (32B/lane -> 2x b128)
__device__ __forceinline__ HFrag load_b_frag(const unsigned int* wf, int tile, int lane) {
  HFrag b;
  const u32x4* p = (const u32x4*)(wf + (size_t)tile * 256) + lane * 2;
  b.q[0] = p[0];
  b.q[1] = p[1];
  return b;
}

__device__ __forceinline__ v8f wmma16(const HFrag& a, const HFrag& b, v8f c) {
  return __builtin_amdgcn_wmma_f32_16x16x32_f16(false, a.h, false, b.h, (short)0, c, false, false);
}

// ---------------- weight prep: fp32 -> f16 B-fragments, zero padded ----------------
__global__ void prep_w_kernel(const float* __restrict__ w, unsigned int* __restrict__ out,
                              int Korig, int Norig, int Kc, int Nt) {
  int idx = blockIdx.x * blockDim.x + threadIdx.x;
  int total = Kc * Nt * 256;
  if (idx >= total) return;
  int tile = idx >> 8, r = idx & 255;
  int lane = r >> 3, v = r & 7;
  int kc = tile / Nt, nt = tile % Nt;
  int h = lane >> 4;
  int n = nt * 16 + (lane & 15);
  int k0 = kc * 32 + b_kin(v, h);
  float x0 = (k0 < Korig && n < Norig) ? w[(size_t)k0 * Norig + n] : 0.f;
  float x1 = (k0 + 1 < Korig && n < Norig) ? w[(size_t)(k0 + 1) * Norig + n] : 0.f;
  out[idx] = pack2h(x0, x1);
}

__global__ void zero_kernel(u32x4* __restrict__ p, long n) {
  long i = blockIdx.x * (long)blockDim.x + threadIdx.x;
  if (i < n) {
    union { unsigned int u[4]; u32x4 q; } z;
    z.u[0] = z.u[1] = z.u[2] = z.u[3] = 0u;
    p[i] = z.q;
  }
}

__global__ void cvt_kernel(const float* __restrict__ in, _Float16* __restrict__ out, int n) {
  int i = blockIdx.x * blockDim.x + threadIdx.x;
  if (i < n) out[i] = (_Float16)in[i];
}

// ---------------- fused point MLP (4->32->64->128->300) + segment max ----------------
#define PT_WAVES 8
__global__ __launch_bounds__(256) void point_mlp_kernel(
    const float* __restrict__ features, const float* __restrict__ coords,
    const int* __restrict__ keypoints, const int* __restrict__ setidx,
    const unsigned int* __restrict__ wsw,
    const float* __restrict__ pb1, const float* __restrict__ pb2,
    const float* __restrict__ pb3, const float* __restrict__ pb4,
    float* __restrict__ agg) {
  __shared__ unsigned int swei[5632];              // w1f + w2f + w3f, 22.5 KB
  __shared__ _Float16 sstage[PT_WAVES][16 * 128];  // per-wave activation stage, 32 KB

  int tid = threadIdx.x;

  // kick off async LDS fill of layer-1..3 weight fragments (overlaps gathers below)
  {
    unsigned sb = lds_off(&swei[0]);
    for (int i = tid; i < 5632 / 4; i += 256)        // 1408 x 16B
      async_copy_b128(sb + i * 16, i * 16, wsw);
  }
  // warm L2/L0 for layer-4 fragment stream (76 KB, reused by all 8 waves)
  for (int i = tid; i < 608; i += 256)
    __builtin_prefetch((const char*)(wsw + W4F_DW) + i * 128, 0, 0);

  int wave = tid >> 5, lane = tid & 31;
  int ln = lane & 15, h = lane >> 4;
  _Float16* stage = sstage[wave];
  int tile = blockIdx.x * PT_WAVES + wave;
  int rowBase = tile * 16;

  // gather inputs and build layer-1 A fragment directly (K=4 zero-padded to 32)
  int row = rowBase + ln;
  int pt = setidx[row];
  int st = setidx[N_PAIRS_C + row];
  int kp = keypoints[st];
  float f0 = features[pt];
  float dx = coords[pt * 3 + 0] - coords[kp * 3 + 0];
  float dy = coords[pt * 3 + 1] - coords[kp * 3 + 1];
  float dz = coords[pt * 3 + 2] - coords[kp * 3 + 2];
  HFrag a1;
#pragma unroll
  for (int v = 0; v < 8; ++v) a1.u[v] = 0u;
  if (h == 0) { a1.u[0] = pack2h(f0, dx); a1.u[1] = pack2h(dy, dz); }

  int setv[8];
#pragma unroll
  for (int v = 0; v < 8; ++v) setv[v] = setidx[N_PAIRS_C + rowBase + v + 8 * h];

  wait_async0();      // weight fragments resident in LDS
  __syncthreads();

  // ---- layer 1: [16x32(K pad)] x [32x32] -> 16x32 ----
#pragma unroll
  for (int nt = 0; nt < 2; ++nt) {
    HFrag b = load_b_frag(swei + W1F_DW, nt, lane);
    CFrag c;
#pragma unroll
    for (int v = 0; v < 8; ++v) c.f[v] = 0.f;
    c.v = wmma16(a1, b, c.v);
    float bb = pb1[nt * 16 + ln];
#pragma unroll
    for (int v = 0; v < 8; ++v) {
      int rm = v + 8 * h;
      stage[rm * 128 + nt * 16 + ln] = (_Float16)fmaxf(c.f[v] + bb, 0.f);
    }
  }

  // ---- layer 2: 16x32 x 32x64 -> 16x64 ----
  {
    HFrag a2 = load_a_frag_lds(stage, 128, 0, lane);
#pragma unroll
    for (int nt = 0; nt < 4; ++nt) {
      HFrag b = load_b_frag(swei + W2F_DW, nt, lane);
      CFrag c;
#pragma unroll
      for (int v = 0; v < 8; ++v) c.f[v] = 0.f;
      c.v = wmma16(a2, b, c.v);
      float bb = pb2[nt * 16 + ln];
#pragma unroll
      for (int v = 0; v < 8; ++v) {
        int rm = v + 8 * h;
        stage[rm * 128 + nt * 16 + ln] = (_Float16)fmaxf(c.f[v] + bb, 0.f);
      }
    }
  }

  // ---- layer 3: 16x64 x 64x128 -> 16x128 ----
  {
    HFrag a3[2];
#pragma unroll
    for (int kc = 0; kc < 2; ++kc) a3[kc] = load_a_frag_lds(stage, 128, kc * 32, lane);
#pragma unroll
    for (int nt = 0; nt < 8; ++nt) {
      CFrag c;
#pragma unroll
      for (int v = 0; v < 8; ++v) c.f[v] = 0.f;
#pragma unroll
      for (int kc = 0; kc < 2; ++kc) {
        HFrag b = load_b_frag(swei + W3F_DW, kc * 8 + nt, lane);
        c.v = wmma16(a3[kc], b, c.v);
      }
      float bb = pb3[nt * 16 + ln];
#pragma unroll
      for (int v = 0; v < 8; ++v) {
        int rm = v + 8 * h;
        stage[rm * 128 + nt * 16 + ln] = (_Float16)fmaxf(c.f[v] + bb, 0.f);
      }
    }
  }

  // ---- layer 4: 16x128 x 128x304 -> 16x304, scatter atomic-max into agg ----
  {
    HFrag a4[4];
#pragma unroll
    for (int kc = 0; kc < 4; ++kc) a4[kc] = load_a_frag_lds(stage, 128, kc * 32, lane);

    for (int nt = 0; nt < 19; ++nt) {
      CFrag c;
#pragma unroll
      for (int v = 0; v < 8; ++v) c.f[v] = 0.f;
#pragma unroll
      for (int kc = 0; kc < 4; ++kc) {
        HFrag b = load_b_frag(wsw + W4F_DW, kc * 19 + nt, lane);
        c.v = wmma16(a4[kc], b, c.v);
      }
      int col = nt * 16 + ln;
      float bb = (col < 300) ? pb4[col] : 0.f;
#pragma unroll
      for (int v = 0; v < 8; ++v) {
        float val = fmaxf(c.f[v] + bb, 0.f);   // ReLU -> val >= 0, int-bit max is valid
        atomicMax((int*)(agg + (size_t)setv[v] * AGG_W + col), __float_as_int(val));
      }
    }
  }
}

// ---------------- output MLP GEMM: [M,320]f16 x [320,304] -> relu, f16 or final f32 ----------------
#define OG_WAVES 4
__global__ __launch_bounds__(128) void out_mlp_kernel(
    const _Float16* __restrict__ Ain, const unsigned int* __restrict__ wfrag,
    const float* __restrict__ bias, _Float16* __restrict__ outH,
    float* __restrict__ outF) {
  __shared__ _Float16 sstage[OG_WAVES][16 * AGG_W];  // 40 KB
  int tid = threadIdx.x, wave = tid >> 5, lane = tid & 31;
  int ln = lane & 15, h = lane >> 4;
  int mt = blockIdx.x * OG_WAVES + wave;
  int rowBase = mt * 16;

  // async LDS fill: this wave's contiguous 16x320 f16 block (10 KB)
  {
    unsigned sb = lds_off(&sstage[wave][0]);
    const void* src = (const void*)(Ain + (size_t)rowBase * AGG_W);
    for (int i = lane; i < 640; i += 32)             // 640 x 16B
      async_copy_b128(sb + i * 16, i * 16, src);
  }
  // warm L2/L0 for the weight fragment stream (190 KB, reused by all waves)
  for (int i = tid; i < 1520; i += 128)
    __builtin_prefetch((const char*)wfrag + i * 128, 0, 0);

  wait_async0();
  __syncthreads();

  HFrag a[10];
#pragma unroll
  for (int kc = 0; kc < 10; ++kc) a[kc] = load_a_frag_lds(sstage[wave], AGG_W, kc * 32, lane);

  for (int nt = 0; nt < 19; ++nt) {
    CFrag c;
#pragma unroll
    for (int v = 0; v < 8; ++v) c.f[v] = 0.f;
#pragma unroll
    for (int kc = 0; kc < 10; ++kc) {
      HFrag b = load_b_frag(wfrag, kc * 19 + nt, lane);
      c.v = wmma16(a[kc], b, c.v);
    }
    int col = nt * 16 + ln;
    float bb = (col < 300) ? bias[col] : 0.f;
#pragma unroll
    for (int v = 0; v < 8; ++v) {
      int rm = v + 8 * h;
      float val = fmaxf(c.f[v] + bb, 0.f);
      if (outF) {
        if (col < 300) outF[(size_t)(rowBase + rm) * 300 + col] = val;
      } else {
        outH[(size_t)(rowBase + rm) * AGG_W + col] = (_Float16)val;
      }
    }
  }
}

extern "C" void kernel_launch(void* const* d_in, const int* in_sizes, int n_in,
                              void* d_out, int out_size, void* d_ws, size_t ws_size,
                              hipStream_t stream) {
  const float* features = (const float*)d_in[0];
  const float* coords   = (const float*)d_in[1];
  const int*   keypoints = (const int*)d_in[2];
  const int*   setidx    = (const int*)d_in[3];
  const float* pw1 = (const float*)d_in[4],  *pb1 = (const float*)d_in[5];
  const float* pw2 = (const float*)d_in[6],  *pb2 = (const float*)d_in[7];
  const float* pw3 = (const float*)d_in[8],  *pb3 = (const float*)d_in[9];
  const float* pw4 = (const float*)d_in[10], *pb4 = (const float*)d_in[11];
  const float* ow1 = (const float*)d_in[12], *ob1 = (const float*)d_in[13];
  const float* ow2 = (const float*)d_in[14], *ob2 = (const float*)d_in[15];

  char* ws = (char*)d_ws;
  unsigned int* wsw = (unsigned int*)ws;
  float*    agg32 = (float*)(ws + AGG32_BYTE);
  _Float16* agg16 = (_Float16*)(ws + AGG16_BYTE);
  _Float16* y1    = (_Float16*)(ws + Y1_BYTE);

  // zero agg (segment-max baseline 0) + f16 buffers (K-padding columns must stay 0)
  long nz = (long)(ZERO_BYTES / 16);
  zero_kernel<<<(int)((nz + 255) / 256), 256, 0, stream>>>((u32x4*)(ws + AGG32_BYTE), nz);

  // weight fragment prep (tiny; runs every call -> deterministic)
  struct PW { const float* w; int off, Ko, No, Kc, Nt; } pws[6] = {
      {pw1, W1F_DW, 4, 32, 1, 2},    {pw2, W2F_DW, 32, 64, 1, 4},
      {pw3, W3F_DW, 64, 128, 2, 8},  {pw4, W4F_DW, 128, 300, 4, 19},
      {ow1, O1F_DW, 300, 300, 10, 19}, {ow2, O2F_DW, 300, 300, 10, 19}};
  for (int i = 0; i < 6; ++i) {
    int total = pws[i].Kc * pws[i].Nt * 256;
    prep_w_kernel<<<(total + 255) / 256, 256, 0, stream>>>(
        pws[i].w, wsw + pws[i].off, pws[i].Ko, pws[i].No, pws[i].Kc, pws[i].Nt);
  }

  // fused point MLP + scatter-max: 32768 row-tiles, 8 per block
  point_mlp_kernel<<<N_PAIRS_C / (16 * PT_WAVES), 256, 0, stream>>>(
      features, coords, keypoints, setidx, wsw, pb1, pb2, pb3, pb4, agg32);

  int ncv = N_KEYPOINTS_C * AGG_W;
  cvt_kernel<<<(ncv + 255) / 256, 256, 0, stream>>>(agg32, agg16, ncv);

  // output MLP: two 300x300 layers (padded 320x304)
  out_mlp_kernel<<<N_KEYPOINTS_C / (16 * OG_WAVES), 128, 0, stream>>>(
      agg16, wsw + O1F_DW, ob1, y1, nullptr);
  out_mlp_kernel<<<N_KEYPOINTS_C / (16 * OG_WAVES), 128, 0, stream>>>(
      y1, wsw + O2F_DW, ob2, nullptr, (float*)d_out);
}
